// SAN_21028159881449
// MI455X (gfx1250) — matrix operand order repaired and checked by reading, not compiled, not run on hardware
//
#include <hip/hip_runtime.h>

#define HEADS  8
#define DHEAD  64
#define NTOK   4096
#define DMODEL 512
#define BATCH  2
#define SCALE  0.125f  /* 64^-0.5 */

typedef __bf16 bf16;
typedef bf16  v16bf __attribute__((ext_vector_type(16)));
typedef bf16  v8bf  __attribute__((ext_vector_type(8)));
typedef float v8f   __attribute__((ext_vector_type(8)));

__device__ __forceinline__ bf16 f2bf(float f) {
    union { float f; unsigned u; } v; v.f = f;
    unsigned r = v.u + 0x7FFFu + ((v.u >> 16) & 1u);   // RNE
    unsigned short h = (unsigned short)(r >> 16);
    bf16 o; __builtin_memcpy(&o, &h, 2); return o;
}

__device__ __forceinline__ v8bf ld8(const bf16* p) { return *(const v8bf*)p; }

__device__ __forceinline__ v16bf cat8(v8bf a, v8bf b) {
    v16bf r;
#pragma unroll
    for (int i = 0; i < 8; ++i) { r[i] = a[i]; r[i + 8] = b[i]; }
    return r;
}

__device__ __forceinline__ v8f wmma_bf(v16bf a, v16bf b, v8f c) {
    return __builtin_amdgcn_wmma_f32_16x16x32_bf16(false, a, false, b,
                                                   (short)0, c, false, false);
}

// CDNA5 async global->LDS copy, 16B per active lane, tracked by ASYNCcnt.
// LDS address operand = low 32 bits of the flat shared pointer (ISA 10.2).
__device__ __forceinline__ void async_cp16(bf16* lds_dst, const bf16* gsrc) {
    unsigned           l = (unsigned)(uintptr_t)lds_dst;
    unsigned long long g = (unsigned long long)(uintptr_t)gsrc;
    asm volatile("global_load_async_to_lds_b128 %0, %1, off"
                 :: "v"(l), "v"(g) : "memory");
}
__device__ __forceinline__ void wait_async0() {
    asm volatile("s_wait_asynccnt 0x0" ::: "memory");
}

// ---------------------------------------------------------------- prep ----
__global__ void san_cvt_bf16(const float* __restrict__ src, bf16* __restrict__ dst, int n) {
    int i = blockIdx.x * blockDim.x + threadIdx.x;
    if (i < n) dst[i] = f2bf(src[i]);
}

// dst[n][k] = (bf16) src[k][n]   (512x512)
__global__ void san_transp_bf16(const float* __restrict__ src, bf16* __restrict__ dst) {
    int i = blockIdx.x * blockDim.x + threadIdx.x;
    int n = i >> 9, k = i & 511;
    dst[i] = f2bf(src[k * DMODEL + n]);
}

// ------------------------------------------------- GEMM1: W = x @ w_qkv ----
// A: [8192,512] bf16 row-major.  Bt: [512 out][512 in] bf16.
// Writes W  [b][h][n][64] and Wt [b][h][64][n] (both bf16).
__global__ void __launch_bounds__(128)
san_gemm_qkv(const bf16* __restrict__ A, const bf16* __restrict__ Bt,
             bf16* __restrict__ W, bf16* __restrict__ Wt) {
    const int tid = threadIdx.x, wv = tid >> 5, lane = tid & 31;
    const int ln = lane & 15, hi = lane >> 4;
    const int row0 = blockIdx.x * 64 + wv * 16;
    const int col0 = blockIdx.y * 64;

    v8f acc[4] = {};
    for (int k0 = 0; k0 < DMODEL; k0 += 32) {
        const bf16* pa = A + (size_t)(row0 + ln) * DMODEL + k0 + hi * 8;
        v16bf a = cat8(ld8(pa), ld8(pa + 16));
#pragma unroll
        for (int nt = 0; nt < 4; ++nt) {
            const bf16* pb = Bt + (size_t)(col0 + nt * 16 + ln) * DMODEL + k0 + hi * 8;
            acc[nt] = wmma_bf(a, cat8(ld8(pb), ld8(pb + 16)), acc[nt]);
        }
    }
#pragma unroll
    for (int nt = 0; nt < 4; ++nt)
#pragma unroll
        for (int r = 0; r < 8; ++r) {
            int t = row0 + r + 8 * hi;
            int c = col0 + nt * 16 + ln;
            int b = t >> 12, i = t & 4095, h = c >> 6, d = c & 63;
            bf16 v = f2bf(acc[nt][r]);
            W [(((size_t)b * HEADS + h) * NTOK + i) * DHEAD + d] = v;
            Wt[(((size_t)b * HEADS + h) * DHEAD + d) * NTOK  + i] = v;
        }
}

// ------------------------------------------------- flash attention --------
// grid (64, 16), block 128.  W=[bh][n][64], Wt=[bh][64][n], O=[b][n][512] bf16
// Double-buffered K/V tiles staged with GLOBAL_LOAD_ASYNC_TO_LDS_B128.
__global__ void __launch_bounds__(128)
san_attn(const bf16* __restrict__ W, const bf16* __restrict__ Wt, bf16* __restrict__ O) {
    __shared__ alignas(16) bf16 Klds[2][64 * 64];     // [buf][key][d]
    __shared__ alignas(16) bf16 Vlds[2][64 * 64];     // [buf][d][key]
    __shared__ alignas(16) bf16 Plds[4][16 * 64];     // per-wave P tile

    const int tid = threadIdx.x, wv = tid >> 5, lane = tid & 31;
    const int ln = lane & 15, hi = lane >> 4;
    const int bh = blockIdx.y;
    const int q0 = blockIdx.x * 64 + wv * 16;

    const bf16* Wh  = W  + (size_t)bh * NTOK * DHEAD;
    const bf16* Wth = Wt + (size_t)bh * DHEAD * NTOK;

    // issue async stage of tile kb into LDS buffer buf (8KB K + 8KB Vt)
    auto stage = [&](int kb, int buf) {
#pragma unroll
        for (int it = 0; it < 4; ++it) {              // K tile: contiguous 8KB
            int c = tid + it * 128;
            async_cp16(&Klds[buf][c * 8], Wh + (size_t)kb * 64 * DHEAD + c * 8);
        }
#pragma unroll
        for (int it = 0; it < 4; ++it) {              // Vt tile: 64 rows x 128B
            int c = tid + it * 128;
            int dr = c >> 3, w8 = c & 7;
            async_cp16(&Vlds[buf][c * 8], Wth + (size_t)dr * NTOK + kb * 64 + w8 * 8);
        }
    };

    v16bf Qa[2];
#pragma unroll
    for (int kc = 0; kc < 2; ++kc) {
        const bf16* p = Wh + (size_t)(q0 + ln) * DHEAD + kc * 32 + hi * 8;
        Qa[kc] = cat8(ld8(p), ld8(p + 16));
    }

    v8f Oacc[4] = {};
    float m[8], l[8];
#pragma unroll
    for (int r = 0; r < 8; ++r) { m[r] = -3.0e38f; l[r] = 0.f; }

    stage(0, 0);

    for (int kb = 0; kb < NTOK / 64; ++kb) {
        const int cur = kb & 1;
        wait_async0();          // my async loads for buf[cur] have landed
        __syncthreads();        // everyone's landed; buf[cur^1] free for reuse

        if (kb + 1 < NTOK / 64) stage(kb + 1, cur ^ 1);  // overlap with compute

        const bf16* Kb = Klds[cur];
        const bf16* Vb = Vlds[cur];

        // S = (Q K^T) * scale : 4 tiles of 16 keys
        v8f S[4];
#pragma unroll
        for (int nt = 0; nt < 4; ++nt) {
            v8f a = {};
#pragma unroll
            for (int kc = 0; kc < 2; ++kc) {
                const bf16* p = Kb + (nt * 16 + ln) * DHEAD + kc * 32 + hi * 8;
                a = wmma_bf(Qa[kc], cat8(ld8(p), ld8(p + 16)), a);
            }
            S[nt] = a * SCALE;
        }

        // online softmax (rows live in VGPR index r + 8*hi; cols across 16 lanes)
        float fac[8], rs[8];
#pragma unroll
        for (int r = 0; r < 8; ++r) {
            float v = fmaxf(fmaxf(S[0][r], S[1][r]), fmaxf(S[2][r], S[3][r]));
#pragma unroll
            for (int msk = 1; msk < 16; msk <<= 1) v = fmaxf(v, __shfl_xor(v, msk, 32));
            float mn = fmaxf(m[r], v);
            fac[r] = __expf(m[r] - mn);
            m[r] = mn;
            rs[r] = 0.f;
        }
#pragma unroll
        for (int nt = 0; nt < 4; ++nt)
#pragma unroll
            for (int r = 0; r < 8; ++r) {
                float p = __expf(S[nt][r] - m[r]);
                rs[r] += p;
                Plds[wv][(r + 8 * hi) * 64 + nt * 16 + ln] = f2bf(p);
            }
#pragma unroll
        for (int r = 0; r < 8; ++r) {
            float v = rs[r];
#pragma unroll
            for (int msk = 1; msk < 16; msk <<= 1) v += __shfl_xor(v, msk, 32);
            l[r] = l[r] * fac[r] + v;
        }
#pragma unroll
        for (int t = 0; t < 4; ++t)
#pragma unroll
            for (int r = 0; r < 8; ++r) Oacc[t][r] *= fac[r];

        // O += P @ V   (per-wave LDS ops are in-order -> no barrier needed)
        v16bf Pa[2];
#pragma unroll
        for (int kc = 0; kc < 2; ++kc) {
            const bf16* p = &Plds[wv][ln * 64 + kc * 32 + hi * 8];
            Pa[kc] = cat8(ld8(p), ld8(p + 16));
        }
#pragma unroll
        for (int dt = 0; dt < 4; ++dt)
#pragma unroll
            for (int kc = 0; kc < 2; ++kc) {
                const bf16* p = Vb + (dt * 16 + ln) * 64 + kc * 32 + hi * 8;
                Oacc[dt] = wmma_bf(Pa[kc], cat8(ld8(p), ld8(p + 16)), Oacc[dt]);
            }
    }

    const int b = bh >> 3, h = bh & 7;
#pragma unroll
    for (int r = 0; r < 8; ++r) {
        float inv = 1.0f / l[r];
        int tok = q0 + r + 8 * hi;
        bf16* orow = O + ((size_t)b * NTOK + tok) * DMODEL + h * DHEAD;
#pragma unroll
        for (int dt = 0; dt < 4; ++dt)
            orow[dt * 16 + ln] = f2bf(Oacc[dt][r] * inv);
    }
}

// ------------------------------------------- GEMM2: Y = O @ w_out + b -----
__global__ void __launch_bounds__(128)
san_gemm_out(const bf16* __restrict__ A, const bf16* __restrict__ Bt,
             const float* __restrict__ bias, float* __restrict__ Y) {
    const int tid = threadIdx.x, wv = tid >> 5, lane = tid & 31;
    const int ln = lane & 15, hi = lane >> 4;
    const int row0 = blockIdx.x * 64 + wv * 16;
    const int col0 = blockIdx.y * 64;

    v8f acc[4] = {};
    for (int k0 = 0; k0 < DMODEL; k0 += 32) {
        const bf16* pa = A + (size_t)(row0 + ln) * DMODEL + k0 + hi * 8;
        v16bf a = cat8(ld8(pa), ld8(pa + 16));
#pragma unroll
        for (int nt = 0; nt < 4; ++nt) {
            const bf16* pb = Bt + (size_t)(col0 + nt * 16 + ln) * DMODEL + k0 + hi * 8;
            acc[nt] = wmma_bf(a, cat8(ld8(pb), ld8(pb + 16)), acc[nt]);
        }
    }
#pragma unroll
    for (int nt = 0; nt < 4; ++nt)
#pragma unroll
        for (int r = 0; r < 8; ++r) {
            int t = row0 + r + 8 * hi;
            int c = col0 + nt * 16 + ln;
            Y[(size_t)t * DMODEL + c] = acc[nt][r] + bias[c];
        }
}

// ----------------------------------------------------------------- launch -
extern "C" void kernel_launch(void* const* d_in, const int* in_sizes, int n_in,
                              void* d_out, int out_size, void* d_ws, size_t ws_size,
                              hipStream_t stream) {
    (void)in_sizes; (void)n_in; (void)out_size; (void)ws_size;
    const float* x     = (const float*)d_in[0];
    const float* w_qkv = (const float*)d_in[1];
    const float* w_out = (const float*)d_in[2];
    const float* b_out = (const float*)d_in[3];

    char* ws = (char*)d_ws;
    size_t off = 0;
    bf16* Xbf  = (bf16*)(ws + off); off += (size_t)BATCH * NTOK * DMODEL * 2;
    bf16* QkvT = (bf16*)(ws + off); off += (size_t)DMODEL * DMODEL * 2;
    bf16* OutT = (bf16*)(ws + off); off += (size_t)DMODEL * DMODEL * 2;
    bf16* W    = (bf16*)(ws + off); off += (size_t)BATCH * HEADS * NTOK * DHEAD * 2;
    bf16* Wt   = (bf16*)(ws + off); off += (size_t)BATCH * HEADS * NTOK * DHEAD * 2;
    bf16* Obf  = (bf16*)(ws + off); off += (size_t)BATCH * NTOK * DMODEL * 2;

    const int nx = BATCH * NTOK * DMODEL;
    san_cvt_bf16<<<(nx + 255) / 256, 256, 0, stream>>>(x, Xbf, nx);
    san_transp_bf16<<<(DMODEL * DMODEL) / 256, 256, 0, stream>>>(w_qkv, QkvT);
    san_transp_bf16<<<(DMODEL * DMODEL) / 256, 256, 0, stream>>>(w_out, OutT);

    san_gemm_qkv<<<dim3(BATCH * NTOK / 64, DMODEL / 64), 128, 0, stream>>>(Xbf, QkvT, W, Wt);
    san_attn    <<<dim3(NTOK / 64, BATCH * HEADS),        128, 0, stream>>>(W, Wt, Obf);
    san_gemm_out<<<dim3(BATCH * NTOK / 64, DMODEL / 64),  128, 0, stream>>>(Obf, OutT, b_out, (float*)d_out);
}
